// LeakyRNN_25039659336106
// MI455X (gfx1250) — matrix operand order repaired and checked by reading, not compiled, not run on hardware
//
#include <hip/hip_runtime.h>
#include <hip/hip_bf16.h>

// ---------------- problem constants (match reference) ----------------
#define T_STEPS 1024
#define BATCH   256
#define HID     256
#define DIM_Y   3
#define WB_IN_STRIDE  260   // DIM_S + DIM_C + DIM_HID + 1 = 2+1+256+1
#define WB_OUT_STRIDE 257   // DIM_HID + 1
#define ALPHA 0.5f

#define NTHREADS 512        // 16 wave32 waves -> 4 waves/SIMD, ~256 VGPRs/wave

typedef __attribute__((ext_vector_type(16))) __bf16 v16bf;
typedef __attribute__((ext_vector_type(8)))  __bf16 v8bf;
typedef __attribute__((ext_vector_type(8)))  float  v8f;
typedef __attribute__((ext_vector_type(4)))  float  v4f;

// ---------------- LDS layout (dynamic shared, 320KB/WGP budget) ------
// [OFF_WH, +128K)  W_h bf16 [256][256] row-major  (W_h[n][k])
// [OFF_HB, +128K)  h = tanh(state) bf16 [256][256] row-major (h[batch][k])
// [OFF_WO, +3K)    W_o f32 [3][256]
// [OFF_BO, +12)    b_o f32 [3]
// [OFF_WI, +4K)    per-hid input-proj consts f32 [256][4] = {Wi0,Wi1,Wi2,b_in}
#define OFF_WH 0
#define OFF_HB (OFF_WH + HID*HID*2)
#define OFF_WO (OFF_HB + BATCH*HID*2)
#define OFF_BO (OFF_WO + DIM_Y*HID*4)
#define OFF_WI (OFF_BO + 16)
#define SMEM_BYTES (OFF_WI + HID*4*4 + 256)   // ~269.5 KB < 320 KB

__device__ __forceinline__ float fast_tanh(float x) {
#if __has_builtin(__builtin_amdgcn_tanhf)
    return __builtin_amdgcn_tanhf(x);          // gfx1250 v_tanh_f32
#elif __has_builtin(__builtin_amdgcn_tanh_f32)
    return __builtin_amdgcn_tanh_f32(x);
#else
    return tanhf(x);
#endif
}

// One persistent workgroup: 16 wave32 waves on one WGP. Recurrent state in
// VGPRs, W_h and tanh(state) resident in LDS, bf16 WMMA on the critical path.
// Each wave owns batch tile bm=wave and all 16 n-tiles, processed as 4 groups
// of 4 independent accumulator chains so WMMA issue overlaps LDS loads.
__global__ __launch_bounds__(NTHREADS)
void leaky_rnn_persistent(const float* __restrict__ s,       // (T,B,2)
                          const float* __restrict__ c,       // (T,B,1)
                          const float* __restrict__ Wb_in,   // (256,260)
                          const float* __restrict__ Wb_out,  // (3,257)
                          float* __restrict__ outputs,       // (T,B,3)
                          float* __restrict__ states)        // (T,B,256)
{
    extern __shared__ char smem[];
    __bf16* Wh = (__bf16*)(smem + OFF_WH);
    __bf16* Hb = (__bf16*)(smem + OFF_HB);
    float*  Wo = (float*) (smem + OFF_WO);
    float*  bo = (float*) (smem + OFF_BO);
    float*  Wi = (float*) (smem + OFF_WI);

    const int tid  = threadIdx.x;
    const int wave = tid >> 5;    // 0..15 == batch tile bm
    const int lane = tid & 31;
    const int ln   = lane & 15;   // N / column within tile
    const int lh   = lane >> 4;   // half-wave select
    const int bm   = wave;        // batch tile 0..15

    // ---- one-time staging of weights into LDS ----
    #pragma unroll 4
    for (int i = 0; i < (HID*HID)/NTHREADS; ++i) {
        int idx = tid + i * NTHREADS;
        int row = idx >> 8, col = idx & 255;
        Wh[idx] = (__bf16)Wb_in[row * WB_IN_STRIDE + 3 + col];  // W_h
        Hb[idx] = (__bf16)0.0f;                                 // tanh(state0)=0
    }
    for (int idx = tid; idx < DIM_Y * HID; idx += NTHREADS)
        Wo[idx] = Wb_out[(idx >> 8) * WB_OUT_STRIDE + (idx & 255)];
    if (tid < DIM_Y) bo[tid] = Wb_out[tid * WB_OUT_STRIDE + HID];
    if (tid < HID) {
        const float* r = Wb_in + (size_t)tid * WB_IN_STRIDE;
        Wi[tid*4+0] = r[0];                  // W_i[:,0] (s0)
        Wi[tid*4+1] = r[1];                  // W_i[:,1] (s1)
        Wi[tid*4+2] = r[2];                  // W_i[:,2] (c)
        Wi[tid*4+3] = r[WB_IN_STRIDE - 1];   // b_in
    }
    __syncthreads();

    // Recurrent state: 16 C/D tiles (f32, WMMA C layout) per wave, in VGPRs.
    v8f state[16];
    #pragma unroll
    for (int j = 0; j < 16; ++j)
        state[j] = (v8f){0.f,0.f,0.f,0.f,0.f,0.f,0.f,0.f};

    // Per-lane invariant LDS addresses
    const __bf16* arow = Hb + (bm * 16 + ln) * HID + lh * 16;  // A row base
    const int abatch = bm * 16 + 8 * lh;                        // first M row

    for (int t = 0; t < T_STEPS; ++t) {
        // ---- per-step inputs for this lane's 8 batch rows ----
        float s0[8], s1[8], cc[8];
        #pragma unroll
        for (int r = 0; r < 8; ++r) {
            long base = (long)t * BATCH + (abatch + r);
            float2 sv = *(const float2*)(s + base * 2);
            s0[r] = sv.x; s1[r] = sv.y;
            cc[r] = c[base];
        }

        // ---- recurrent GEMM: acc = tanh(state_prev) @ W_h^T  (bf16 WMMA) ----
        // 4 groups of 4 n-tiles: kt-outer so the A fragment is loaded once per
        // (group, kt) and the 4 inner WMMAs are independent chains.
        #pragma unroll
        for (int g = 0; g < 4; ++g) {
            v8f acc[4];
            #pragma unroll
            for (int jj = 0; jj < 4; ++jj)
                acc[jj] = (v8f){0.f,0.f,0.f,0.f,0.f,0.f,0.f,0.f};

            #pragma unroll
            for (int kt = 0; kt < 8; ++kt) {
                // A fragment (16x32 bf16): lane row M=bm*16+ln,
                // K = kt*32 + lh*16 + [0..15]
                v8bf alo = *(const v8bf*)(arow + kt * 32);
                v8bf ahi = *(const v8bf*)(arow + kt * 32 + 8);
                v16bf A = __builtin_shufflevector(alo, ahi,
                          0,1,2,3,4,5,6,7,8,9,10,11,12,13,14,15);
                #pragma unroll
                for (int jj = 0; jj < 4; ++jj) {
                    const int hid = (g * 4 + jj) * 16 + ln;
                    // B fragment (32x16 bf16): lane holds column n=hid,
                    // K = kt*32 + lh*16 + [0..15] -> 32 contiguous bytes
                    v16bf B = *(const v16bf*)(Wh + hid * HID + kt * 32 + lh * 16);
                    acc[jj] = __builtin_amdgcn_wmma_f32_16x16x32_bf16(
                                  false, A, false, B, (short)0, acc[jj],
                                  false, false);
                }
            }

            // ---- leaky update: state = (1-a)*state + a*(x + acc) ----
            #pragma unroll
            for (int jj = 0; jj < 4; ++jj) {
                const int j   = g * 4 + jj;
                const int hid = j * 16 + ln;
                const float wi0 = Wi[hid*4+0], wi1 = Wi[hid*4+1];
                const float wi2 = Wi[hid*4+2], bi  = Wi[hid*4+3];
                v8f st = state[j];
                #pragma unroll
                for (int r = 0; r < 8; ++r) {
                    int batch = abatch + r;
                    float x  = fmaf(s0[r], wi0,
                               fmaf(s1[r], wi1, fmaf(cc[r], wi2, bi)));
                    float sn = (1.0f - ALPHA) * st[r] + ALPHA * (x + acc[jj][r]);
                    st[r] = sn;
                    states[((long)t * BATCH + batch) * HID + hid] = sn;
                }
                state[j] = st;
            }
        }

        __syncthreads();   // all readers of old Hb (GEMM + prev out-proj) done

        // ---- write h = tanh(state_new) back to LDS ----
        #pragma unroll
        for (int j = 0; j < 16; ++j) {
            const int hid = j * 16 + ln;
            #pragma unroll
            for (int r = 0; r < 8; ++r) {
                int batch = abatch + r;
                Hb[batch * HID + hid] = (__bf16)fast_tanh(state[j][r]);
            }
        }

        __syncthreads();   // new Hb visible to everyone

        // ---- output head: out[t,b,y] = h[b,:] . W_o[y,:] + b_o[y] ----
        #pragma unroll
        for (int idx = tid; idx < BATCH * DIM_Y; idx += NTHREADS) {
            const int b = idx / 3, y = idx - 3 * b;
            const float* wrow = Wo + y * HID;
            const __bf16* hrow = Hb + b * HID;
            float accv = bo[y];
            #pragma unroll 8
            for (int k = 0; k < HID; k += 8) {
                v8bf h8 = *(const v8bf*)(hrow + k);
                v4f  w0 = *(const v4f*)(wrow + k);
                v4f  w1 = *(const v4f*)(wrow + k + 4);
                accv = fmaf((float)h8[0], w0[0], accv);
                accv = fmaf((float)h8[1], w0[1], accv);
                accv = fmaf((float)h8[2], w0[2], accv);
                accv = fmaf((float)h8[3], w0[3], accv);
                accv = fmaf((float)h8[4], w1[0], accv);
                accv = fmaf((float)h8[5], w1[1], accv);
                accv = fmaf((float)h8[6], w1[2], accv);
                accv = fmaf((float)h8[7], w1[3], accv);
            }
            outputs[((long)t * BATCH + b) * DIM_Y + y] = accv;
        }
        // Hb is only overwritten after every thread passes the next barrier #1,
        // so these reads are safe w.r.t. the next iteration.
    }
}

extern "C" void kernel_launch(void* const* d_in, const int* in_sizes, int n_in,
                              void* d_out, int out_size, void* d_ws, size_t ws_size,
                              hipStream_t stream) {
    (void)in_sizes; (void)n_in; (void)out_size; (void)d_ws; (void)ws_size;
    const float* s      = (const float*)d_in[0];
    const float* c      = (const float*)d_in[1];
    const float* Wb_in  = (const float*)d_in[2];
    const float* Wb_out = (const float*)d_in[3];
    float* outputs = (float*)d_out;                              // (T,B,3)
    float* states  = outputs + (long)T_STEPS * BATCH * DIM_Y;    // (T,B,256)

    // >64KB dynamic LDS (CDNA5 WGP has 320KB): opt in explicitly.
    hipFuncSetAttribute((const void*)leaky_rnn_persistent,
                        hipFuncAttributeMaxDynamicSharedMemorySize, SMEM_BYTES);

    leaky_rnn_persistent<<<dim3(1), dim3(NTHREADS), SMEM_BYTES, stream>>>(
        s, c, Wb_in, Wb_out, outputs, states);
}
